// GroverFinetuneTask_for_PP_9964324127446
// MI455X (gfx1250) — matrix operand order repaired and checked by reading, not compiled, not run on hardware
//
#include <hip/hip_runtime.h>
#include <hip/hip_bf16.h>

#define NA   20000
#define NBD  40000
#define MAXNB 6
#define HID  256
#define NHEAD 4

typedef __attribute__((ext_vector_type(16))) __bf16 bf16x16;
typedef __attribute__((ext_vector_type(8)))  float  f32x8;

#define GF_RELU 1
#define GF_BIAS 2
#define GF_RES  4

#define TM 128
#define TN 64
#define TK 32
#define LSTR 40   // LDS row stride in bf16: 80B rows -> 16B-aligned b128 async
                  // destinations; 20-dword step walks 16 distinct banks.

__device__ __forceinline__ unsigned short f2bf(float f) {
  unsigned int u = __float_as_uint(f);
  u += 0x7FFFu + ((u >> 16) & 1u);   // round-to-nearest-even
  return (unsigned short)(u >> 16);
}

// ---------------------------------------------------------------------------
// Weight prep: Wt[n][kp] = bf16(W[k][n]), K zero-padded to Kp (multiple of 32).
// Lets the GEMM stage B tiles as contiguous bf16 rows via async LDS DMA.
// ---------------------------------------------------------------------------
__global__ __launch_bounds__(256) void wt_kernel(
    const float* __restrict__ W, int K, int N, int Kp,
    unsigned short* __restrict__ Wt)
{
  int n = blockIdx.x;
  for (int k = threadIdx.x; k < Kp; k += blockDim.x)
    Wt[(size_t)n * Kp + k] =
        (k < K) ? f2bf(W[(size_t)k * N + n]) : (unsigned short)0;
}

// ---------------------------------------------------------------------------
// Tiled GEMM: C[M,N] = act( concat(A1,A2)[M,K] @ W[K,N] (+bias) (+Res) )
// B operand is pre-transposed bf16 Wt[N][Kp]; tiles staged to LDS with
// GLOBAL_LOAD_ASYNC_TO_LDS_B128 (ASYNCcnt). A staged f32->bf16; float4 fast
// path when contiguous. bf16 WMMA, f32 accumulate.
// ---------------------------------------------------------------------------
__global__ __launch_bounds__(256) void gemm_kernel(
    const float* __restrict__ A1, int lda1,
    const float* __restrict__ A2, int lda2, int K1,
    const unsigned short* __restrict__ Wt, int Kp,
    const float* __restrict__ bias,
    const float* __restrict__ Res, int ldres,
    float* __restrict__ C, int ldc,
    int M, int N, int K, int flags, int fastA)
{
  __shared__ __align__(16) unsigned short As[TM * LSTR];
  __shared__ __align__(16) unsigned short Bs[TN * LSTR];

  const int tid  = threadIdx.x;
  const int lane = tid & 31;
  const int wid  = tid >> 5;       // 8 waves
  const int wm   = wid & 3;        // 4 waves along M
  const int wn   = wid >> 2;       // 2 waves along N
  const int m0   = blockIdx.x * TM;
  const int n0   = blockIdx.y * TN;

  // Per-thread async-copy slot for the B tile: 64 rows x 4 x 16B.
  const int brow = tid >> 2;       // 0..63
  const int bseg = tid & 3;        // 0..3
  const unsigned int bs_lds =
      (unsigned int)(uintptr_t)&Bs[brow * LSTR + bseg * 8];
  const unsigned short* bsrc =
      Wt + (size_t)(n0 + brow) * Kp + bseg * 8;

  const bool fast = fastA && (m0 + TM <= M);

  f32x8 acc[2][2];
  for (int i = 0; i < 2; ++i)
    for (int j = 0; j < 2; ++j)
      for (int r = 0; r < 8; ++r) acc[i][j][r] = 0.0f;

  for (int k0 = 0; k0 < K; k0 += TK) {
    // ---- B tile: one async b128 DMA per lane, zero VALU staging ----
    asm volatile("global_load_async_to_lds_b128 %0, %1, off"
                 :: "v"(bs_lds), "v"(bsrc + k0) : "memory");

    // ---- A tile (128 x 32), f32 -> bf16 ----
    if (fast) {
      for (int i = tid; i < (TM * TK) / 4; i += 256) {
        int row = i >> 3, c4 = (i & 7) << 2;
        const float4 f4 = *reinterpret_cast<const float4*>(
            &A1[(size_t)(m0 + row) * lda1 + k0 + c4]);
        uint2 p;
        p.x = ((unsigned int)f2bf(f4.y) << 16) | f2bf(f4.x);
        p.y = ((unsigned int)f2bf(f4.w) << 16) | f2bf(f4.z);
        *reinterpret_cast<uint2*>(&As[row * LSTR + c4]) = p;
      }
      if (k0 + TK < K)   // pull next A tile toward the caches
        __builtin_prefetch(&A1[(size_t)(m0 + (tid >> 1)) * lda1 + k0 + TK], 0, 1);
    } else {
      for (int i = tid; i < TM * TK; i += 256) {
        int row = i >> 5, kk = i & 31;
        int gm = m0 + row, gk = k0 + kk;
        float v = 0.0f;
        if (gm < M && gk < K)
          v = (gk < K1) ? A1[(size_t)gm * lda1 + gk]
                        : A2[(size_t)gm * lda2 + (gk - K1)];
        As[row * LSTR + kk] = f2bf(v);
      }
    }
    asm volatile("s_wait_asynccnt 0x0" ::: "memory");
    __syncthreads();

    // ---- wave fragments per ISA 7.12.2 layouts ----
    union Frag { unsigned int u[8]; bf16x16 v; } af[2], bfr[2];
    const int khalfA = (lane & 16) ? 8 : 0;
    const int khalfB = (lane & 16) ? 8 : 0;
    for (int fm = 0; fm < 2; ++fm) {
      int mrow = wm * 32 + fm * 16 + (lane & 15);
      const unsigned short* ap = &As[mrow * LSTR];
      for (int d = 0; d < 8; ++d) {
        int k = ((d & 4) ? 16 : 0) + khalfA + ((d & 3) << 1);
        af[fm].u[d] = *reinterpret_cast<const unsigned int*>(ap + k);
      }
    }
    for (int fn = 0; fn < 2; ++fn) {
      int ncol = wn * 32 + fn * 16 + (lane & 15);
      const unsigned short* bp = &Bs[ncol * LSTR];
      for (int d = 0; d < 8; ++d) {
        int k = ((d & 4) ? 16 : 0) + khalfB + ((d & 3) << 1);
        bfr[fn].u[d] = *reinterpret_cast<const unsigned int*>(bp + k);
      }
    }
    for (int fm = 0; fm < 2; ++fm)
      for (int fn = 0; fn < 2; ++fn)
        acc[fm][fn] = __builtin_amdgcn_wmma_f32_16x16x32_bf16(
            false, af[fm].v, false, bfr[fn].v, (short)0, acc[fm][fn],
            false, false);
    __syncthreads();
  }

  // ---- fused epilogue ----
  for (int fm = 0; fm < 2; ++fm)
    for (int fn = 0; fn < 2; ++fn)
      for (int r = 0; r < 8; ++r) {
        int gm = m0 + wm * 32 + fm * 16 + r + ((lane & 16) ? 8 : 0);
        int gn = n0 + wn * 32 + fn * 16 + (lane & 15);
        if (gm < M && gn < N) {
          float v = acc[fm][fn][r];
          if (flags & GF_BIAS) v += bias[gn];
          if (flags & GF_RES)  v += Res[(size_t)gm * ldres + gn];
          if (flags & GF_RELU) v = fmaxf(v, 0.0f);
          C[(size_t)gm * ldc + gn] = v;
        }
      }
}

// ---------------------------------------------------------------------------
// LayerNorm over 256 features: one wave32 per row, shuffle reductions.
// ---------------------------------------------------------------------------
__global__ __launch_bounds__(256) void ln_kernel(
    const float* __restrict__ X, const float* __restrict__ g,
    const float* __restrict__ b, float* __restrict__ Y, int M)
{
  int gw = (blockIdx.x * 256 + threadIdx.x) >> 5;
  int lane = threadIdx.x & 31;
  if (gw >= M) return;
  const float4* x = reinterpret_cast<const float4*>(X + (size_t)gw * HID) + lane * 2;
  float4 v0 = x[0], v1 = x[1];
  float s  = v0.x + v0.y + v0.z + v0.w + v1.x + v1.y + v1.z + v1.w;
  float s2 = v0.x*v0.x + v0.y*v0.y + v0.z*v0.z + v0.w*v0.w
           + v1.x*v1.x + v1.y*v1.y + v1.z*v1.z + v1.w*v1.w;
  for (int off = 16; off >= 1; off >>= 1) {
    s  += __shfl_xor(s,  off, 32);
    s2 += __shfl_xor(s2, off, 32);
  }
  float mean = s * (1.0f / HID);
  float var  = s2 * (1.0f / HID) - mean * mean;
  float inv  = rsqrtf(var + 1e-5f);
  const float4* gp = reinterpret_cast<const float4*>(g) + lane * 2;
  const float4* bp = reinterpret_cast<const float4*>(b) + lane * 2;
  float4* y = reinterpret_cast<float4*>(Y + (size_t)gw * HID) + lane * 2;
  for (int i = 0; i < 2; ++i) {
    float4 vv = (i == 0) ? v0 : v1, gg = gp[i], bb = bp[i], o;
    o.x = (vv.x - mean) * inv * gg.x + bb.x;
    o.y = (vv.y - mean) * inv * gg.y + bb.y;
    o.z = (vv.z - mean) * inv * gg.z + bb.z;
    o.w = (vv.w - mean) * inv * gg.w + bb.w;
    y[i] = o;
  }
}

// dst[m,:] = sum_j src[idx[m*J+j], :]   (64 threads = one float4 lane each)
__global__ __launch_bounds__(64) void gather_sum_kernel(
    const float* __restrict__ src, const int* __restrict__ idx, int J,
    float* __restrict__ dst)
{
  int m = blockIdx.x, d4 = threadIdx.x;
  float4 s = {0.f, 0.f, 0.f, 0.f};
  for (int j = 0; j < J; ++j) {
    const float4 v = *reinterpret_cast<const float4*>(
        &src[(size_t)idx[m * J + j] * HID + d4 * 4]);
    s.x += v.x; s.y += v.y; s.z += v.z; s.w += v.w;
  }
  *reinterpret_cast<float4*>(&dst[(size_t)m * HID + d4 * 4]) = s;
}

// dst[b,:] = nei[b2a[b],:] - msg[b2revb[b],:]
__global__ __launch_bounds__(64) void bond_combine_kernel(
    const float* __restrict__ nei, const float* __restrict__ msg,
    const int* __restrict__ b2a, const int* __restrict__ b2revb,
    float* __restrict__ dst)
{
  int b = blockIdx.x, d4 = threadIdx.x;
  const float4 n4 = *reinterpret_cast<const float4*>(
      &nei[(size_t)b2a[b] * HID + d4 * 4]);
  const float4 m4 = *reinterpret_cast<const float4*>(
      &msg[(size_t)b2revb[b] * HID + d4 * 4]);
  float4 o = {n4.x - m4.x, n4.y - m4.y, n4.z - m4.z, n4.w - m4.w};
  *reinterpret_cast<float4*>(&dst[(size_t)b * HID + d4 * 4]) = o;
}

// ba[b,:] = ah[b2a[b]] + sum_j ah[a2a[b2a[b],j]] - ah[b2a[b2revb[b]]]
__global__ __launch_bounds__(64) void ba_kernel(
    const float* __restrict__ ah, const int* __restrict__ a2a,
    const int* __restrict__ b2a, const int* __restrict__ b2revb,
    float* __restrict__ dst)
{
  int b = blockIdx.x, d4 = threadIdx.x;
  int a0 = b2a[b];
  float4 s = *reinterpret_cast<const float4*>(&ah[(size_t)a0 * HID + d4 * 4]);
  for (int j = 0; j < MAXNB; ++j) {
    const float4 v = *reinterpret_cast<const float4*>(
        &ah[(size_t)a2a[a0 * MAXNB + j] * HID + d4 * 4]);
    s.x += v.x; s.y += v.y; s.z += v.z; s.w += v.w;
  }
  const float4 r = *reinterpret_cast<const float4*>(
      &ah[(size_t)b2a[b2revb[b]] * HID + d4 * 4]);
  s.x -= r.x; s.y -= r.y; s.z -= r.z; s.w -= r.w;
  *reinterpret_cast<float4*>(&dst[(size_t)b * HID + d4 * 4]) = s;
}

// bb[b,:] = sum_j bh[a2b[b2a[b],j]] - bh[b2revb[b]]
__global__ __launch_bounds__(64) void bb_kernel(
    const float* __restrict__ bh, const int* __restrict__ a2b,
    const int* __restrict__ b2a, const int* __restrict__ b2revb,
    float* __restrict__ dst)
{
  int b = blockIdx.x, d4 = threadIdx.x;
  int a0 = b2a[b];
  float4 s = {0.f, 0.f, 0.f, 0.f};
  for (int j = 0; j < MAXNB; ++j) {
    const float4 v = *reinterpret_cast<const float4*>(
        &bh[(size_t)a2b[a0 * MAXNB + j] * HID + d4 * 4]);
    s.x += v.x; s.y += v.y; s.z += v.z; s.w += v.w;
  }
  const float4 r = *reinterpret_cast<const float4*>(
      &bh[(size_t)b2revb[b] * HID + d4 * 4]);
  s.x -= r.x; s.y -= r.y; s.z -= r.z; s.w -= r.w;
  *reinterpret_cast<float4*>(&dst[(size_t)b * HID + d4 * 4]) = s;
}

// ---------------------------------------------------------------------------
// Per-node 4x4 attention: one wave32 per (node, mha-head).
// Row layout: [(n*4 + headmodule)*256 + mha*64 + d]. O may alias Q.
// ---------------------------------------------------------------------------
__global__ __launch_bounds__(256) void attn_kernel(
    const float* __restrict__ Q, const float* __restrict__ Kk,
    const float* __restrict__ V, float* __restrict__ O, int Nn)
{
  int gw = (blockIdx.x * 256 + threadIdx.x) >> 5;
  int lane = threadIdx.x & 31;
  int n = gw >> 2, mh = gw & 3;
  if (n >= Nn) return;
  float2 q[4], k[4], v[4];
  size_t base0 = ((size_t)n * 4) * HID + mh * 64 + lane * 2;
  for (int s = 0; s < 4; ++s) {
    size_t b = base0 + (size_t)s * HID;
    q[s] = *reinterpret_cast<const float2*>(Q  + b);
    k[s] = *reinterpret_cast<const float2*>(Kk + b);
    v[s] = *reinterpret_cast<const float2*>(V  + b);
  }
  float S[4][4];
  for (int s = 0; s < 4; ++s)
    for (int t = 0; t < 4; ++t) {
      float p = q[s].x * k[t].x + q[s].y * k[t].y;
      for (int off = 16; off >= 1; off >>= 1) p += __shfl_xor(p, off, 32);
      S[s][t] = p * 0.125f;  // 1/sqrt(64)
    }
  float2 o[4];
  for (int s = 0; s < 4; ++s) {
    float mx = fmaxf(fmaxf(S[s][0], S[s][1]), fmaxf(S[s][2], S[s][3]));
    float e0 = __expf(S[s][0] - mx), e1 = __expf(S[s][1] - mx);
    float e2 = __expf(S[s][2] - mx), e3 = __expf(S[s][3] - mx);
    float inv = 1.0f / (e0 + e1 + e2 + e3);
    o[s].x = (e0 * v[0].x + e1 * v[1].x + e2 * v[2].x + e3 * v[3].x) * inv;
    o[s].y = (e0 * v[0].y + e1 * v[1].y + e2 * v[2].y + e3 * v[3].y) * inv;
  }
  for (int s = 0; s < 4; ++s)
    *reinterpret_cast<float2*>(O + base0 + (size_t)s * HID) = o[s];
}

// ============================ host orchestration ===========================

struct BlkW {
  const float *W_i, *ln1_g, *ln1_b, *Wh_q, *Wh_k, *Wh_v,
              *Wq, *Wk, *Wv, *Wao, *W_o, *ln2_g, *ln2_b;
};
struct FfnW { const float *W1, *b1, *W2, *b2, *ln_g, *ln_b; };

struct WT { const unsigned short* w; int Kp; };
struct BlkWT { WT Wi; WT Wh[3][4]; WT Wp[3]; WT Wao; WT Wo; };
struct FfnWT { WT W1, W2; };

static inline void launch_gemm(hipStream_t st,
    const float* A1, int lda1, const float* A2, int lda2, int K1,
    WT B, const float* bias, const float* Res, int ldres,
    float* C, int ldc, int M, int N, int K, int flags)
{
  int fastA = (A2 == nullptr) && ((lda1 & 3) == 0) && ((K & 31) == 0);
  dim3 g((M + TM - 1) / TM, (N + TN - 1) / TN);
  gemm_kernel<<<g, dim3(256), 0, st>>>(
      A1, lda1, A2 ? A2 : A1, lda2, A2 ? K1 : K, B.w, B.Kp,
      bias ? bias : A1, Res ? Res : A1, ldres, C, ldc, M, N, K, flags, fastA);
}

static inline void gemmN(hipStream_t st, const float* A, int lda, WT B,
                         float* C, int ldc, int M, int N, int K, int flags,
                         const float* bias = nullptr,
                         const float* Res = nullptr, int ldres = 0)
{
  launch_gemm(st, A, lda, nullptr, 0, K, B, bias, Res, ldres,
              C, ldc, M, N, K, flags);
}

static void run_mtblock(hipStream_t st, const float* f, int fdim, int Mm,
                        bool atom_msgs, const BlkW& P, const BlkWT& T,
                        const int* a2nei, const int* b2a, const int* b2revb,
                        float* out_h, char* wsBase, size_t& wsOff)
{
  auto walloc = [&](size_t n) -> float* {
    float* p = (float*)(wsBase + wsOff);
    wsOff = (wsOff + n * 4 + 255) & ~(size_t)255;
    return p;
  };
  float* x   = walloc((size_t)Mm * HID);
  float* t1  = walloc((size_t)Mm * HID);
  float* agg = walloc((size_t)Mm * HID);
  float* nei = walloc((size_t)NA * HID);
  float* QP  = walloc((size_t)Mm * 4 * HID);
  float* KP  = walloc((size_t)Mm * 4 * HID);
  float* VP  = walloc((size_t)Mm * 4 * HID);

  // x = LN(ReLU(f @ W_i))
  gemmN(st, f, fdim, T.Wi, x, HID, Mm, HID, fdim, GF_RELU);
  ln_kernel<<<(Mm + 7) / 8, 256, 0, st>>>(x, P.ln1_g, P.ln1_b, x, Mm);

  float* slab[3] = {QP, KP, VP};

  for (int h = 0; h < NHEAD; ++h)
    for (int c = 0; c < 3; ++c) {
      for (int hop = 0; hop < 2; ++hop) {          // DEPTH-1 = 2 hops
        const float* msg_in = (hop == 0) ? x : t1;
        if (atom_msgs) {
          gather_sum_kernel<<<Mm, 64, 0, st>>>(msg_in, a2nei, MAXNB, agg);
        } else {
          gather_sum_kernel<<<NA, 64, 0, st>>>(msg_in, a2nei, MAXNB, nei);
          bond_combine_kernel<<<Mm, 64, 0, st>>>(nei, msg_in, b2a, b2revb, agg);
        }
        // t1 = ReLU(x + agg @ Wh)
        gemmN(st, agg, HID, T.Wh[c][h], t1, HID, Mm, HID, HID,
              GF_RELU | GF_RES, nullptr, x, HID);
      }
      // project this head-slice directly into the [Mm,4,256] slab
      gemmN(st, t1, HID, T.Wp[c], slab[c] + h * HID, 4 * HID,
            Mm, HID, HID, 0);
    }

  int waves = Mm * 4;
  attn_kernel<<<(waves + 7) / 8, 256, 0, st>>>(QP, KP, VP, QP, Mm);
  gemmN(st, QP, HID, T.Wao, KP, HID, 4 * Mm, HID, HID, 0);
  gemmN(st, KP, 4 * HID, T.Wo, out_h, HID, Mm, HID, 4 * HID, 0);
  ln_kernel<<<(Mm + 7) / 8, 256, 0, st>>>(out_h, P.ln2_g, P.ln2_b, out_h, Mm);
}

static void run_ffn(hipStream_t st, const float* f, int fdim, const float* agg,
                    const FfnW& P, const FfnWT& T, float* out, int M,
                    char* wsBase, size_t& wsOff)
{
  float* hidden = (float*)(wsBase + wsOff);
  wsOff = (wsOff + (size_t)M * 4 * HID * 4 + 255) & ~(size_t)255;
  // hidden = ReLU(concat(f, agg) @ W1 + b1)
  launch_gemm(st, f, fdim, agg, HID, fdim, T.W1, P.b1, nullptr, 0,
              hidden, 4 * HID, M, 4 * HID, fdim + HID, GF_RELU | GF_BIAS);
  gemmN(st, hidden, 4 * HID, T.W2, out, HID, M, HID, 4 * HID,
        GF_BIAS, P.b2);
  ln_kernel<<<(M + 7) / 8, 256, 0, st>>>(out, P.ln_g, P.ln_b, out, M);
}

extern "C" void kernel_launch(void* const* d_in, const int* in_sizes, int n_in,
                              void* d_out, int out_size, void* d_ws,
                              size_t ws_size, hipStream_t stream)
{
  (void)in_sizes; (void)n_in; (void)out_size; (void)ws_size;

  int pi = 0;
  const float* f_atoms = (const float*)d_in[pi++];
  const float* f_bonds = (const float*)d_in[pi++];
  auto getB = [&]() {
    BlkW b;
    b.W_i  = (const float*)d_in[pi++]; b.ln1_g = (const float*)d_in[pi++];
    b.ln1_b= (const float*)d_in[pi++]; b.Wh_q  = (const float*)d_in[pi++];
    b.Wh_k = (const float*)d_in[pi++]; b.Wh_v  = (const float*)d_in[pi++];
    b.Wq   = (const float*)d_in[pi++]; b.Wk    = (const float*)d_in[pi++];
    b.Wv   = (const float*)d_in[pi++]; b.Wao   = (const float*)d_in[pi++];
    b.W_o  = (const float*)d_in[pi++]; b.ln2_g = (const float*)d_in[pi++];
    b.ln2_b= (const float*)d_in[pi++];
    return b;
  };
  BlkW node = getB(), edge = getB();
  auto getF = [&]() {
    FfnW f;
    f.W1  = (const float*)d_in[pi++]; f.b1   = (const float*)d_in[pi++];
    f.W2  = (const float*)d_in[pi++]; f.b2   = (const float*)d_in[pi++];
    f.ln_g= (const float*)d_in[pi++]; f.ln_b = (const float*)d_in[pi++];
    return f;
  };
  FfnW faa = getF(), fab = getF(), fba = getF(), fbb = getF();
  const int* a2b    = (const int*)d_in[pi++];
  const int* a2a    = (const int*)d_in[pi++];
  const int* b2a    = (const int*)d_in[pi++];
  const int* b2revb = (const int*)d_in[pi++];

  float* out    = (float*)d_out;
  float* out_aa = out;
  float* out_ab = out + (size_t)NA * HID;
  float* out_ba = out + (size_t)2 * NA * HID;
  float* out_bb = out + (size_t)2 * NA * HID + (size_t)NBD * HID;

  char* wsBase = (char*)d_ws;
  size_t wsOff = 0;
  auto walloc_b = [&](size_t bytes) -> void* {
    void* p = (void*)(wsBase + wsOff);
    wsOff = (wsOff + bytes + 255) & ~(size_t)255;
    return p;
  };

  // ---- persistent bf16 transposed weight pool ----
  auto prep = [&](const float* W, int K, int N) -> WT {
    int Kp = (K + 31) & ~31;
    unsigned short* wt = (unsigned short*)walloc_b((size_t)N * Kp * 2);
    wt_kernel<<<N, 256, 0, stream>>>(W, K, N, Kp, wt);
    WT r; r.w = wt; r.Kp = Kp; return r;
  };
  auto prepBlock = [&](const BlkW& b, int fdim) -> BlkWT {
    BlkWT t;
    t.Wi = prep(b.W_i, fdim, HID);
    const float* Whs[3] = {b.Wh_q, b.Wh_k, b.Wh_v};
    for (int c = 0; c < 3; ++c)
      for (int h = 0; h < NHEAD; ++h)
        t.Wh[c][h] = prep(Whs[c] + (size_t)h * HID * HID, HID, HID);
    t.Wp[0] = prep(b.Wq, HID, HID);
    t.Wp[1] = prep(b.Wk, HID, HID);
    t.Wp[2] = prep(b.Wv, HID, HID);
    t.Wao   = prep(b.Wao, HID, HID);
    t.Wo    = prep(b.W_o, 4 * HID, HID);
    return t;
  };
  auto prepFfn = [&](const FfnW& f, int fdim) -> FfnWT {
    FfnWT t;
    t.W1 = prep(f.W1, fdim + HID, 4 * HID);
    t.W2 = prep(f.W2, 4 * HID, HID);
    return t;
  };

  BlkWT nodeT = prepBlock(node, 151), edgeT = prepBlock(edge, 165);
  FfnWT faaT = prepFfn(faa, 151), fabT = prepFfn(fab, 151);
  FfnWT fbaT = prepFfn(fba, 165), fbbT = prepFfn(fbb, 165);

  float* atoms_h = (float*)walloc_b((size_t)NA * HID * 4);
  float* bonds_h = (float*)walloc_b((size_t)NBD * HID * 4);
  size_t mark = wsOff;

  // ---- MTBlocks ----
  run_mtblock(stream, f_atoms, 151, NA,  true,  node, nodeT, a2a, b2a, b2revb,
              atoms_h, wsBase, wsOff);
  wsOff = mark;
  run_mtblock(stream, f_bonds, 165, NBD, false, edge, edgeT, a2b, b2a, b2revb,
              bonds_h, wsBase, wsOff);
  wsOff = mark;

  // ---- output FFNs ----
  float* agg = (float*)walloc_b((size_t)NBD * HID * 4);  // sized for bonds
  size_t mark2 = wsOff;

  gather_sum_kernel<<<NA, 64, 0, stream>>>(atoms_h, a2a, MAXNB, agg);
  wsOff = mark2;
  run_ffn(stream, f_atoms, 151, agg, faa, faaT, out_aa, NA, wsBase, wsOff);

  gather_sum_kernel<<<NA, 64, 0, stream>>>(bonds_h, a2b, MAXNB, agg);
  wsOff = mark2;
  run_ffn(stream, f_atoms, 151, agg, fab, fabT, out_ab, NA, wsBase, wsOff);

  ba_kernel<<<NBD, 64, 0, stream>>>(atoms_h, a2a, b2a, b2revb, agg);
  wsOff = mark2;
  run_ffn(stream, f_bonds, 165, agg, fba, fbaT, out_ba, NBD, wsBase, wsOff);

  bb_kernel<<<NBD, 64, 0, stream>>>(bonds_h, a2b, b2a, b2revb, agg);
  wsOff = mark2;
  run_ffn(stream, f_bonds, 165, agg, fbb, fbbT, out_bb, NBD, wsBase, wsOff);
}
